// LinearCrossAttention_66254165508385
// MI455X (gfx1250) — matrix operand order repaired and checked by reading, not compiled, not run on hardware
//
#include <hip/hip_runtime.h>

// ---------------------------------------------------------------------------
// Problem constants (fixed by the reference)
// ---------------------------------------------------------------------------
#define D_MODEL 1024
#define N_HEADS 16
#define N_KV    4
#define D_HEAD  64
#define N_DOCS  8
#define EPS_F   1e-5f

// ---------------------------------------------------------------------------
// Types (clang ext vectors are trivial -> union-safe)
// ---------------------------------------------------------------------------
typedef __attribute__((ext_vector_type(16))) __bf16        bf16x16;
typedef __attribute__((ext_vector_type(8)))  float         f32x8;
typedef __attribute__((ext_vector_type(4)))  unsigned int  u32x4;
typedef __attribute__((ext_vector_type(8)))  int           i32x8;
typedef __attribute__((ext_vector_type(4)))  int           i32x4;

union FragAB { u32x4 u[2]; bf16x16 v; };

__device__ __forceinline__ unsigned short f2bf(float f) {
    union { float f; unsigned u; } c; c.f = f;
    unsigned r = c.u + 0x7FFFu + ((c.u >> 16) & 1u);   // round-to-nearest-even
    return (unsigned short)(r >> 16);
}

__device__ __forceinline__ float phi_act(float x) {    // elu(x)+1
    return x > 0.f ? x + 1.f : __expf(x);
}

// ---------------------------------------------------------------------------
// CDNA5 feature selection (compile-probe via __has_builtin; #warning -> stderr)
// ---------------------------------------------------------------------------
#if __has_builtin(__builtin_amdgcn_tensor_load_to_lds)
  #define USE_TDM 1
  #warning "CDNA5 path: tensor_load_to_lds (TDM) tile staging selected"
#else
  #define USE_TDM 0
  #warning "CDNA5 path: TDM builtin unavailable -> bf16 VGPR staging fallback"
#endif

template <short N>
__device__ __forceinline__ void wait_tensorcnt() {
#if __has_builtin(__builtin_amdgcn_s_wait_tensorcnt)
    __builtin_amdgcn_s_wait_tensorcnt(N);   // N is a constant expression here
#else
    asm volatile("s_wait_tensorcnt %0" :: "i"(N));
#endif
}

#if USE_TDM
// 2D TDM load: tile = rows x 32 bf16, global row stride = K elements,
// LDS rows padded 64B -> 80B via descriptor padding (matches LDSS=40 shorts).
__device__ __forceinline__ void tdm_load_2d(unsigned lds_off,
                                            const unsigned short* gptr,
                                            int rows, int K)
{
    const unsigned long long ga = (unsigned long long)(size_t)gptr;
    u32x4 g0 = {
        1u,                                            // count=1 (valid, user mode)
        lds_off,                                       // lds_addr [63:32]
        (unsigned)ga,                                  // global_addr lo
        (unsigned)((ga >> 32) & 0x1FFFFFFu) | (2u << 30)  // addr[56:32] | type=2
    };
    i32x8 g1;
    g1[0] = (int)((1u << 16)        // data_size = 2 bytes
                | (1u << 20)        // pad_enable
                | (3u << 22)        // pad_interval: 16 DWORDs (64B)
                | (3u << 25));      // pad_amount:   4 DWORDs (16B)
    g1[1] = (int)((unsigned)(K & 0xFFFF) << 16);                 // tensor_dim0[15:0]
    g1[2] = (int)(((unsigned)K >> 16) | ((unsigned)(rows & 0xFFFF) << 16)); // dim0 hi / dim1 lo
    g1[3] = (int)(32u << 16);                                    // dim1 hi=0 | tile_dim0=32
    g1[4] = rows & 0xFFFF;                                       // tile_dim1 | tile_dim2=0
    g1[5] = K;                                                   // tensor_dim0_stride lo32
    g1[6] = 0;                                                   // stride0 hi16 | stride1 lo16
    g1[7] = 0;
    i32x4 z4 = {0, 0, 0, 0};
#if defined(__clang_major__) && (__clang_major__ >= 23)
    i32x8 z8 = {0, 0, 0, 0, 0, 0, 0, 0};
    __builtin_amdgcn_tensor_load_to_lds(g0, g1, z4, z4, z8, 0);
#else
    __builtin_amdgcn_tensor_load_to_lds(g0, g1, z4, z4, 0);
#endif
}
#endif

// ---------------------------------------------------------------------------
// bulk fp32 -> bf16 conversion (memory-bound one-shot pass)
// ---------------------------------------------------------------------------
__global__ __launch_bounds__(256)
void cvt_f32_bf16(const float* __restrict__ in, unsigned short* __restrict__ out,
                  size_t n)
{
    size_t i = ((size_t)blockIdx.x * 256 + threadIdx.x) * 8;
    if (i >= n) return;
    float4 a = *(const float4*)(in + i);
    float4 b = *(const float4*)(in + i + 4);
    ushort4 lo, hi;
    lo.x = f2bf(a.x); lo.y = f2bf(a.y); lo.z = f2bf(a.z); lo.w = f2bf(a.w);
    hi.x = f2bf(b.x); hi.y = f2bf(b.y); hi.z = f2bf(b.z); hi.w = f2bf(b.w);
    *(ushort4*)(out + i)     = lo;
    *(ushort4*)(out + i + 4) = hi;
}

// ---------------------------------------------------------------------------
// GEMM: C[M,N] = epilogue( A[M,K] @ W[N,K]^T ), A/W bf16, C fp32.
// TDM double-buffered tile staging; WMMA bf16 compute w/ f32 accumulate.
// flags bit0: phi epilogue ; bit1: rowmask[m] multiply
// ---------------------------------------------------------------------------
#define BM 128
#define BN 64
#define BK 32
#define LDSS 40   // padded bf16 row stride: 80 bytes (16B-aligned rows)

__global__ __launch_bounds__(256)
void gemm_bf16_wmma(const unsigned short* __restrict__ A,
                    const unsigned short* __restrict__ W,
                    float* __restrict__ C, int M, int N, int K,
                    int flags, const unsigned char* __restrict__ rowmask)
{
    __shared__ __align__(16) unsigned short sA[2][BM * LDSS];
    __shared__ __align__(16) unsigned short sB[2][BN * LDSS];

    const int tid  = threadIdx.x;
    const int m0   = blockIdx.x * BM;
    const int n0   = blockIdx.y * BN;

    const int wave = tid >> 5;
    const int lane = tid & 31;
    const int half = lane >> 4;
    const int l16  = lane & 15;
    const int wm0  = (wave & 3) * 32;   // 4 waves along M
    const int wn0  = (wave >> 2) * 32;  // 2 waves along N

    f32x8 acc[2][2];
#pragma unroll
    for (int i = 0; i < 2; ++i)
#pragma unroll
        for (int j = 0; j < 2; ++j)
#pragma unroll
            for (int e = 0; e < 8; ++e) acc[i][j][e] = 0.f;

    const int kTiles = K / BK;

#if USE_TDM
    // wave 0 drives the Tensor Data Mover; TENSORcnt is per-wave.
    if (wave == 0) {
        tdm_load_2d((unsigned)(size_t)(const void*)&sA[0][0],
                    A + (size_t)m0 * K, BM, K);
        tdm_load_2d((unsigned)(size_t)(const void*)&sB[0][0],
                    W + (size_t)n0 * K, BN, K);
    }
#endif

    for (int kt = 0; kt < kTiles; ++kt) {
        const int buf = kt & 1;
#if USE_TDM
        if (wave == 0) {
            if (kt + 1 < kTiles) {
                const int k1 = (kt + 1) * BK;
                tdm_load_2d((unsigned)(size_t)(const void*)&sA[buf ^ 1][0],
                            A + (size_t)m0 * K + k1, BM, K);
                tdm_load_2d((unsigned)(size_t)(const void*)&sB[buf ^ 1][0],
                            W + (size_t)n0 * K + k1, BN, K);
                wait_tensorcnt<2>();   // tile kt complete, next tile in flight
            } else {
                wait_tensorcnt<0>();
            }
        }
        __syncthreads();
#else
        {   // fallback: bf16 tiles through VGPRs (no conversion -> few VALU)
            const int k0 = kt * BK;
#pragma unroll
            for (int p = 0; p < 2; ++p) {              // A: 512 x 16B chunks
                int idx = tid + p * 256;
                int r = idx >> 2, c = (idx & 3) << 3;  // 4 chunks of 8 bf16 / row
                *(u32x4*)(&sA[buf][r * LDSS + c]) =
                    *(const u32x4*)(A + (size_t)(m0 + r) * K + k0 + c);
            }
            {                                           // B: 256 x 16B chunks
                int r = tid >> 2, c = (tid & 3) << 3;
                *(u32x4*)(&sB[buf][r * LDSS + c]) =
                    *(const u32x4*)(W + (size_t)(n0 + r) * K + k0 + c);
            }
        }
        __syncthreads();
#endif
        // ---- fragments per ISA 7.12.2 (wave32)
        FragAB fa[2], fb[2];
#pragma unroll
        for (int t = 0; t < 2; ++t) {
            const unsigned short* rp = &sA[buf][(wm0 + t * 16 + l16) * LDSS];
            fa[t].u[0] = *(const u32x4*)(rp + half * 8);
            fa[t].u[1] = *(const u32x4*)(rp + 16 + half * 8);
        }
#pragma unroll
        for (int t = 0; t < 2; ++t) {
            const unsigned short* rp = &sB[buf][(wn0 + t * 16 + l16) * LDSS];
            fb[t].u[0] = *(const u32x4*)(rp + half * 8);
            fb[t].u[1] = *(const u32x4*)(rp + 16 + half * 8);
        }
#pragma unroll
        for (int tm = 0; tm < 2; ++tm)
#pragma unroll
            for (int tn = 0; tn < 2; ++tn)
                acc[tm][tn] = __builtin_amdgcn_wmma_f32_16x16x32_bf16(
                    false, fa[tm].v, false, fb[tn].v,
                    (short)0, acc[tm][tn], false, false);
        __syncthreads();
    }

    // ---- epilogue: C/D layout element (m = half*8 + r, n = l16) per VGPR r
#pragma unroll
    for (int tm = 0; tm < 2; ++tm) {
#pragma unroll
        for (int r = 0; r < 8; ++r) {
            const int m = m0 + wm0 + tm * 16 + half * 8 + r;
            float mk = 1.f;
            if (flags & 2) mk = rowmask[m] ? 1.f : 0.f;
#pragma unroll
            for (int tn = 0; tn < 2; ++tn) {
                const int n = n0 + wn0 + tn * 16 + l16;
                float v = acc[tm][tn][r];
                if (flags & 1) v = phi_act(v);
                C[(size_t)m * N + n] = v * mk;
            }
        }
    }
}

// ---------------------------------------------------------------------------
// Per-document K/V statistics:  ctx[b,kv,d,i,j] = sum_s k_i v_j ; z = sum_s k
// One block per (b, kv). Sorted src_doc_ids -> one contiguous run per doc ->
// register accumulators flushed once per run, no atomics.
// ---------------------------------------------------------------------------
#define CH 64

__global__ __launch_bounds__(256)
void doc_stats(const float* __restrict__ Kb, const float* __restrict__ Vb,
               const int* __restrict__ sdoc, float* __restrict__ ctx,
               float* __restrict__ zvec, int S)
{
    __shared__ __align__(16) float sk[CH][D_HEAD];
    __shared__ __align__(16) float sv[CH][D_HEAD];
    __shared__ int sd[CH];

    const int b   = blockIdx.x >> 2;
    const int kv  = blockIdx.x & 3;
    const int tid = threadIdx.x;
    const int i   = tid & 63;
    const int jg  = tid >> 6;
    const int j0  = jg * 16;

    float acc[16];
#pragma unroll
    for (int jj = 0; jj < 16; ++jj) acc[jj] = 0.f;
    float zacc = 0.f;
    int curdoc = -1;

    const size_t kvcols = (size_t)N_KV * D_HEAD;   // 256
    for (int s0 = 0; s0 < S; s0 += CH) {
#pragma unroll
        for (int p = 0; p < 4; ++p) {
            int idx = tid + p * 256;
            int r   = idx >> 4;
            int q4  = (idx & 15) << 2;
            size_t off = ((size_t)b * S + s0 + r) * kvcols + kv * D_HEAD + q4;
            *(float4*)&sk[r][q4] = *(const float4*)(Kb + off);
            *(float4*)&sv[r][q4] = *(const float4*)(Vb + off);
        }
        if (tid < CH) sd[tid] = sdoc[(size_t)b * S + s0 + tid];
        __syncthreads();

        for (int s = 0; s < CH; ++s) {
            const int d = sd[s];                     // uniform across block
            if (d != curdoc) {
                if (curdoc >= 0) {
                    float* cp = ctx + (((size_t)(b * N_KV + kv) * N_DOCS + curdoc)
                                       * D_HEAD + i) * D_HEAD + j0;
#pragma unroll
                    for (int jj = 0; jj < 16; ++jj) { cp[jj] = acc[jj]; acc[jj] = 0.f; }
                    if (jg == 0)
                        zvec[((size_t)(b * N_KV + kv) * N_DOCS + curdoc) * D_HEAD + i] = zacc;
                    zacc = 0.f;
                }
                curdoc = d;
            }
            const float ki = sk[s][i];
#pragma unroll
            for (int jj = 0; jj < 16; ++jj) acc[jj] += ki * sv[s][j0 + jj];
            if (jg == 0) zacc += ki;
        }
        __syncthreads();
    }
    if (curdoc >= 0) {
        float* cp = ctx + (((size_t)(b * N_KV + kv) * N_DOCS + curdoc)
                           * D_HEAD + i) * D_HEAD + j0;
#pragma unroll
        for (int jj = 0; jj < 16; ++jj) cp[jj] = acc[jj];
        if (jg == 0)
            zvec[((size_t)(b * N_KV + kv) * N_DOCS + curdoc) * D_HEAD + i] = zacc;
    }
}

// ---------------------------------------------------------------------------
// attn[b,l,h*64+j] = (q_h . ctx[:,j]) / (q_h . z + eps), emitted as bf16 so
// the Wo GEMM consumes it directly. ctx (2MB) is L2-resident.
// ---------------------------------------------------------------------------
__global__ __launch_bounds__(256)
void num_den(const float* __restrict__ Q, const float* __restrict__ ctx,
             const float* __restrict__ zvec, const int* __restrict__ tdoc,
             unsigned short* __restrict__ attn, int L)
{
    __shared__ __align__(16) float sq[D_MODEL];
    const int bl  = blockIdx.x;          // b*L + l
    const int tid = threadIdx.x;
    *(float4*)&sq[tid * 4] = *(const float4*)(Q + (size_t)bl * D_MODEL + tid * 4);
    __syncthreads();

    const int d   = tdoc[bl];
    const int h   = tid >> 4;
    const int l16 = tid & 15;
    const int kv  = h >> 2;
    const int b   = bl / L;

    const float* qh = sq + h * D_HEAD;
    const float* cb = ctx + (((size_t)(b * N_KV + kv) * N_DOCS + d) * D_HEAD) * D_HEAD;
    const float* zb = zvec + ((size_t)(b * N_KV + kv) * N_DOCS + d) * D_HEAD;

    float num0 = 0.f, num1 = 0.f, num2 = 0.f, num3 = 0.f, den = 0.f;
#pragma unroll 4
    for (int i = 0; i < D_HEAD; ++i) {
        const float qi = qh[i];
        den += qi * zb[i];
        const float* ci = cb + (size_t)i * D_HEAD + l16;
        num0 += qi * ci[0];
        num1 += qi * ci[16];
        num2 += qi * ci[32];
        num3 += qi * ci[48];
    }
    const float inv = 1.f / (den + EPS_F);
    unsigned short* op = attn + (size_t)bl * D_MODEL + h * D_HEAD + l16;
    op[0]  = f2bf(num0 * inv);
    op[16] = f2bf(num1 * inv);
    op[32] = f2bf(num2 * inv);
    op[48] = f2bf(num3 * inv);
}

// ---------------------------------------------------------------------------
// Launch
// ---------------------------------------------------------------------------
extern "C" void kernel_launch(void* const* d_in, const int* in_sizes, int n_in,
                              void* d_out, int out_size, void* d_ws, size_t ws_size,
                              hipStream_t stream)
{
    (void)in_sizes; (void)n_in; (void)out_size; (void)ws_size;

    const float*         x     = (const float*)d_in[0];
    const float*         enc   = (const float*)d_in[1];
    const unsigned char* emask = (const unsigned char*)d_in[2];
    const int*           sdoc  = (const int*)d_in[3];
    const int*           tdoc  = (const int*)d_in[4];
    // d_in[5] = n_docs (compile-time constant)
    const float*         Wq    = (const float*)d_in[6];
    const float*         Wk    = (const float*)d_in[7];
    const float*         Wv    = (const float*)d_in[8];
    const float*         Wo    = (const float*)d_in[9];
    float*               out   = (float*)d_out;

    const int B = 4, L = 4096, S = 4096;
    const int ML = B * L, MS = B * S;
    const int NKV = N_KV * D_HEAD;          // 256

    char* ws = (char*)d_ws;
    unsigned short* x_bf   = (unsigned short*)ws; ws += (size_t)ML * D_MODEL * 2;
    unsigned short* enc_bf = (unsigned short*)ws; ws += (size_t)MS * D_MODEL * 2;
    unsigned short* Wq_bf  = (unsigned short*)ws; ws += (size_t)D_MODEL * D_MODEL * 2;
    unsigned short* Wk_bf  = (unsigned short*)ws; ws += (size_t)NKV * D_MODEL * 2;
    unsigned short* Wv_bf  = (unsigned short*)ws; ws += (size_t)NKV * D_MODEL * 2;
    unsigned short* Wo_bf  = (unsigned short*)ws; ws += (size_t)D_MODEL * D_MODEL * 2;
    unsigned short* at_bf  = (unsigned short*)ws; ws += (size_t)ML * D_MODEL * 2;
    float* Qb  = (float*)ws; ws += (size_t)ML * D_MODEL * 4;
    float* Kb  = (float*)ws; ws += (size_t)MS * NKV * 4;
    float* Vb  = (float*)ws; ws += (size_t)MS * NKV * 4;
    float* ctx = (float*)ws; ws += (size_t)B * N_KV * N_DOCS * D_HEAD * D_HEAD * 4;
    float* zv  = (float*)ws;

    const size_t statBytes = ((size_t)B * N_KV * N_DOCS * D_HEAD * D_HEAD
                            + (size_t)B * N_KV * N_DOCS * D_HEAD) * sizeof(float);
    (void)hipMemsetAsync(ctx, 0, statBytes, stream);

    dim3 blk(256);
    auto cvtGrid = [](size_t n) { return dim3((unsigned)(n / 8 / 256)); };
    cvt_f32_bf16<<<cvtGrid((size_t)ML * D_MODEL), blk, 0, stream>>>(x,   x_bf,   (size_t)ML * D_MODEL);
    cvt_f32_bf16<<<cvtGrid((size_t)MS * D_MODEL), blk, 0, stream>>>(enc, enc_bf, (size_t)MS * D_MODEL);
    cvt_f32_bf16<<<cvtGrid((size_t)D_MODEL * D_MODEL), blk, 0, stream>>>(Wq, Wq_bf, (size_t)D_MODEL * D_MODEL);
    cvt_f32_bf16<<<cvtGrid((size_t)NKV * D_MODEL), blk, 0, stream>>>(Wk, Wk_bf, (size_t)NKV * D_MODEL);
    cvt_f32_bf16<<<cvtGrid((size_t)NKV * D_MODEL), blk, 0, stream>>>(Wv, Wv_bf, (size_t)NKV * D_MODEL);
    cvt_f32_bf16<<<cvtGrid((size_t)D_MODEL * D_MODEL), blk, 0, stream>>>(Wo, Wo_bf, (size_t)D_MODEL * D_MODEL);

    // Q = phi(x @ Wq^T)
    gemm_bf16_wmma<<<dim3(ML / BM, D_MODEL / BN), blk, 0, stream>>>(
        x_bf, Wq_bf, Qb, ML, D_MODEL, D_MODEL, /*flags=*/1, nullptr);
    // K = phi(enc @ Wk^T) * mask
    gemm_bf16_wmma<<<dim3(MS / BM, NKV / BN), blk, 0, stream>>>(
        enc_bf, Wk_bf, Kb, MS, NKV, D_MODEL, /*flags=*/3, emask);
    // V = (enc @ Wv^T) * mask
    gemm_bf16_wmma<<<dim3(MS / BM, NKV / BN), blk, 0, stream>>>(
        enc_bf, Wv_bf, Vb, MS, NKV, D_MODEL, /*flags=*/2, emask);
    // per-doc ctx / z
    doc_stats<<<dim3(B * N_KV), blk, 0, stream>>>(Kb, Vb, sdoc, ctx, zv, S);
    // normalized attention output (bf16)
    num_den<<<dim3(ML), blk, 0, stream>>>(Qb, ctx, zv, tdoc, at_bf, L);
    // out = attn @ Wo^T
    gemm_bf16_wmma<<<dim3(ML / BM, D_MODEL / BN), blk, 0, stream>>>(
        at_bf, Wo_bf, out, ML, D_MODEL, D_MODEL, /*flags=*/0, nullptr);
}
